// RelativeAttention_34668976013601
// MI455X (gfx1250) — matrix operand order repaired and checked by reading, not compiled, not run on hardware
//
#include <hip/hip_runtime.h>
#include <hip/hip_bf16.h>
#include <stdint.h>

// ---------------- problem constants ----------------
#define Bt   16
#define Ct   384
#define Ht   32
#define Wt   32
#define Nt   (Ht * Wt)        // 1024 spatial positions
#define Mt   (Bt * Nt)        // 16384 total rows
#define OCt  (3 * Ct)         // 1152 qkv output channels

// ---------------- vector types ----------------
typedef __attribute__((ext_vector_type(16))) __bf16 v16bf;
typedef __attribute__((ext_vector_type(8)))  float  v8f;
typedef __attribute__((ext_vector_type(4)))  unsigned int u32x4;
typedef __attribute__((ext_vector_type(4)))  int    v4i;
typedef __attribute__((ext_vector_type(4)))  float  f32x4;

typedef __attribute__((address_space(1))) v4i* gptr_v4i;   // global
typedef __attribute__((address_space(3))) v4i* lptr_v4i;   // LDS

union Frag16 {                  // 16 bf16 = 32 bytes = 8 VGPRs
    u32x4 h[2];
    v16bf v;
};

__device__ __forceinline__ v8f wmma_bf16(v16bf a, v16bf b, v8f c) {
    // D = A(16x32) * B(32x16) + C, f32 accumulate
    return __builtin_amdgcn_wmma_f32_16x16x32_bf16(
        false, a, false, b, (short)0, c, false, false);
}

__device__ __forceinline__ unsigned short f2bf(float f) {
    unsigned int u = __float_as_uint(f);
    u += 0x7FFFu + ((u >> 16) & 1u);     // round-to-nearest-even
    return (unsigned short)(u >> 16);
}

// ---------------- workspace layout (bytes) ----------------
#define OFF_XB  0
#define OFF_WB  (OFF_XB + (size_t)Mt * Ct * 2)
#define OFF_QB  (OFF_WB + (size_t)OCt * Ct * 2)
#define OFF_KF  (OFF_QB + (size_t)Mt * Ct * 2)
#define OFF_KB  (OFF_KF + (size_t)Mt * Ct * 4)
#define OFF_VT  (OFF_KB + (size_t)Mt * Ct * 2)

// ================= kernel 1: w_qkv f32 -> bf16 =================
__global__ void k_cvt_w(const float* __restrict__ w, unsigned short* __restrict__ wb) {
    int i = blockIdx.x * blockDim.x + threadIdx.x;
    if (i < OCt * Ct) wb[i] = f2bf(w[i]);
}

// ========== kernel 2: x [B,C,N] f32 -> x_flat [B*N, C] bf16 ==========
__global__ void k_xpose(const float* __restrict__ x, unsigned short* __restrict__ xb) {
    int i = blockIdx.x * blockDim.x + threadIdx.x;      // over B*C*N
    if (i >= Bt * Ct * Nt) return;
    int b = i / (Ct * Nt);
    int r = i % (Ct * Nt);
    int c = r / Nt;
    int n = r % Nt;
    xb[((size_t)(b * Nt + n)) * Ct + c] = f2bf(x[i]);
}

// ========== kernel 3: qkv GEMM via WMMA ==========
// out[m][o] = sum_c xb[m][c] * wb[o][c] + bias[o]
// WG: 256 thr = 8 waves (4x2), tile 128(M) x 64(O); wave tile 32x32 (2x2 wmma)
__global__ __launch_bounds__(256)
void k_qkv_gemm(const unsigned short* __restrict__ xb,
                const unsigned short* __restrict__ wb,
                const float* __restrict__ bias,
                unsigned short* __restrict__ qb,
                float* __restrict__ kf,
                unsigned short* __restrict__ vT) {
    const int wave = threadIdx.x >> 5;
    const int lane = threadIdx.x & 31;
    const int l15  = lane & 15;
    const int half = lane >> 4;
    const int wm = wave >> 1;           // 0..3
    const int wn = wave & 1;            // 0..1
    const int row0 = blockIdx.y * 128 + wm * 32;
    const int col0 = blockIdx.x * 64  + wn * 32;

    v8f acc[2][2];
    #pragma unroll
    for (int i = 0; i < 2; ++i)
        #pragma unroll
        for (int j = 0; j < 2; ++j) acc[i][j] = (v8f)(0.0f);

    for (int kc = 0; kc < Ct; kc += 32) {
        Frag16 A[2], Bf[2];
        #pragma unroll
        for (int mt = 0; mt < 2; ++mt) {
            const unsigned short* p = xb + ((size_t)(row0 + mt * 16 + l15)) * Ct + kc;
            A[mt].h[0] = *(const u32x4*)(p + half * 8);
            A[mt].h[1] = *(const u32x4*)(p + 16 + half * 8);
        }
        #pragma unroll
        for (int nt = 0; nt < 2; ++nt) {
            const unsigned short* p = wb + ((size_t)(col0 + nt * 16 + l15)) * Ct + kc;
            Bf[nt].h[0] = *(const u32x4*)(p + half * 8);
            Bf[nt].h[1] = *(const u32x4*)(p + 16 + half * 8);
        }
        #pragma unroll
        for (int mt = 0; mt < 2; ++mt)
            #pragma unroll
            for (int nt = 0; nt < 2; ++nt)
                acc[mt][nt] = wmma_bf16(A[mt].v, Bf[nt].v, acc[mt][nt]);
    }

    #pragma unroll
    for (int mt = 0; mt < 2; ++mt) {
        #pragma unroll
        for (int nt = 0; nt < 2; ++nt) {
            const int col = col0 + nt * 16 + l15;
            const float bs = bias[col];
            #pragma unroll
            for (int r = 0; r < 8; ++r) {
                const int m = row0 + mt * 16 + r + 8 * half;
                const float val = acc[mt][nt][r] + bs;
                if (col < Ct) {                          // q -> bf16 [M][C]
                    qb[(size_t)m * Ct + col] = f2bf(val);
                } else if (col < 2 * Ct) {               // k -> f32 [M][C]
                    kf[(size_t)m * Ct + (col - Ct)] = val;
                } else {                                 // v -> bf16 [B][C][N]
                    const int b  = m >> 10;
                    const int ns = m & (Nt - 1);
                    vT[((size_t)(b * Ct + (col - 2 * Ct))) * Nt + ns] = f2bf(val);
                }
            }
        }
    }
}

// ========== kernel 4: depthwise 3x3 conv + bias, fused k' = k + rb ==========
__global__ void k_dwconv_addk(const float* __restrict__ x,
                              const float* __restrict__ wdw,
                              const float* __restrict__ bdw,
                              const float* __restrict__ kf,
                              unsigned short* __restrict__ kb) {
    int i = blockIdx.x * blockDim.x + threadIdx.x;      // over B*C*N
    if (i >= Bt * Ct * Nt) return;
    int b = i / (Ct * Nt);
    int r = i % (Ct * Nt);
    int c = r / Nt;
    int n = r % Nt;
    int h = n / Wt, w = n % Wt;

    const float* xp = x + ((size_t)(b * Ct + c)) * Nt;
    const float* wp = wdw + c * 9;
    float rb = 0.0f;
    #pragma unroll
    for (int dh = 0; dh < 3; ++dh) {
        int hh = h + dh - 1;
        if (hh < 0 || hh >= Ht) continue;
        #pragma unroll
        for (int dw = 0; dw < 3; ++dw) {
            int ww = w + dw - 1;
            if (ww < 0 || ww >= Wt) continue;
            rb += xp[hh * Wt + ww] * wp[dh * 3 + dw];
        }
    }
    rb += bdw[c];
    const size_t m = (size_t)(b * Nt + n);
    kb[m * Ct + c] = f2bf(kf[m * Ct + c] + rb);
}

// ========== kernel 5: fused attention (S -> softmax -> P*V) ==========
// grid: B * (N/32) blocks of 256 threads; S tile 32x1024 lives in LDS.
#define QROWS 32
#define SMEM_Q   ((size_t)QROWS * Ct * 2)        // 24576
#define SMEM_S   ((size_t)QROWS * Nt * 4)        // 131072
#define SMEM_P   ((size_t)QROWS * Nt * 2)        // 65536
#define SMEM_RED ((size_t)256 * 4)               // 1024
#define SMEM_TOTAL (SMEM_Q + SMEM_S + SMEM_P + SMEM_RED)   // 222208 B < 320 KB

__global__ __launch_bounds__(256)
void k_attention(const unsigned short* __restrict__ qb,
                 const unsigned short* __restrict__ kb,
                 const unsigned short* __restrict__ vT,
                 float* __restrict__ out) {
    extern __shared__ unsigned char smem[];
    unsigned short* Qs = (unsigned short*)smem;                       // [32][384] bf16
    float*          Ss = (float*)(smem + SMEM_Q);                     // [32][1024] f32
    unsigned short* Ps = (unsigned short*)(smem + SMEM_Q + SMEM_S);   // [32][1024] bf16
    float*         red = (float*)(smem + SMEM_Q + SMEM_S + SMEM_P);   // [256]

    const int b  = blockIdx.x >> 5;
    const int q0 = (blockIdx.x & 31) * QROWS;
    const int tid  = threadIdx.x;
    const int wave = tid >> 5;
    const int lane = tid & 31;
    const int l15  = lane & 15;
    const int half = lane >> 4;

    // ---- stage Q block (contiguous 32*384 bf16) into LDS ----
#if __has_builtin(__builtin_amdgcn_global_load_async_to_lds_b128)
    {
        const unsigned short* gsrc = qb + ((size_t)(b * Nt + q0)) * Ct;
        for (int i = tid; i < (QROWS * Ct) / 8; i += 256) {
            __builtin_amdgcn_global_load_async_to_lds_b128(
                (gptr_v4i)(gsrc + i * 8),
                (lptr_v4i)(Qs + i * 8),
                0, 0);
        }
    #if __has_builtin(__builtin_amdgcn_s_wait_asynccnt)
        __builtin_amdgcn_s_wait_asynccnt(0);
    #else
        asm volatile("s_wait_asynccnt 0x0" ::: "memory");
    #endif
    }
#else
    {
        const u32x4* src = (const u32x4*)(qb + ((size_t)(b * Nt + q0)) * Ct);
        u32x4* dst = (u32x4*)Qs;
        for (int i = tid; i < (QROWS * Ct) / 8; i += 256) dst[i] = src[i];
    }
#endif
    __syncthreads();

    // ---- phase 1: S = Q * (k+rb)^T ; wave handles 128 key columns ----
    {
        v8f acc[2][8];
        #pragma unroll
        for (int mt = 0; mt < 2; ++mt)
            #pragma unroll
            for (int nt = 0; nt < 8; ++nt) acc[mt][nt] = (v8f)(0.0f);

        for (int kc = 0; kc < Ct; kc += 32) {
            Frag16 A[2];
            #pragma unroll
            for (int mt = 0; mt < 2; ++mt) {
                const unsigned short* p = Qs + (mt * 16 + l15) * Ct + kc;
                A[mt].h[0] = *(const u32x4*)(p + half * 8);
                A[mt].h[1] = *(const u32x4*)(p + 16 + half * 8);
            }
            #pragma unroll
            for (int nt = 0; nt < 8; ++nt) {
                const int key = wave * 128 + nt * 16 + l15;
                const unsigned short* p =
                    kb + ((size_t)(b * Nt + key)) * Ct + kc;
                Frag16 Bf;
                Bf.h[0] = *(const u32x4*)(p + half * 8);
                Bf.h[1] = *(const u32x4*)(p + 16 + half * 8);
                acc[0][nt] = wmma_bf16(A[0].v, Bf.v, acc[0][nt]);
                acc[1][nt] = wmma_bf16(A[1].v, Bf.v, acc[1][nt]);
            }
        }
        #pragma unroll
        for (int mt = 0; mt < 2; ++mt)
            #pragma unroll
            for (int nt = 0; nt < 8; ++nt)
                #pragma unroll
                for (int r = 0; r < 8; ++r)
                    Ss[(mt * 16 + r + 8 * half) * Nt + wave * 128 + nt * 16 + l15] =
                        acc[mt][nt][r];
    }
    __syncthreads();

    // ---- phase 2: row softmax in LDS, vectorized b128 (8 threads/row) ----
    {
        const int row  = tid >> 3;
        const int part = tid & 7;
        f32x4* srow4 = (f32x4*)(Ss + row * Nt + part * 128);

        float mx = -3.4e38f;
        #pragma unroll 4
        for (int j = 0; j < 32; ++j) {
            f32x4 v = srow4[j];
            mx = fmaxf(mx, fmaxf(fmaxf(v.x, v.y), fmaxf(v.z, v.w)));
        }
        red[tid] = mx;
        __syncthreads();
        float rmax = red[row * 8];
        #pragma unroll
        for (int j = 1; j < 8; ++j) rmax = fmaxf(rmax, red[row * 8 + j]);
        __syncthreads();

        float sum = 0.0f;
        #pragma unroll 4
        for (int j = 0; j < 32; ++j) {
            f32x4 v = srow4[j];
            v.x = __expf(v.x - rmax);
            v.y = __expf(v.y - rmax);
            v.z = __expf(v.z - rmax);
            v.w = __expf(v.w - rmax);
            srow4[j] = v;
            sum += v.x + v.y + v.z + v.w;
        }
        red[tid] = sum;
        __syncthreads();
        float rsum = 0.0f;
        #pragma unroll
        for (int j = 0; j < 8; ++j) rsum += red[row * 8 + j];
        const float inv = 1.0f / rsum;

        unsigned int* prow = (unsigned int*)(Ps + row * Nt + part * 128);
        #pragma unroll 4
        for (int j = 0; j < 32; ++j) {
            f32x4 v = srow4[j];
            prow[2 * j]     = (unsigned int)f2bf(v.x * inv) |
                              ((unsigned int)f2bf(v.y * inv) << 16);
            prow[2 * j + 1] = (unsigned int)f2bf(v.z * inv) |
                              ((unsigned int)f2bf(v.w * inv) << 16);
        }
    }
    __syncthreads();

    // ---- phase 3: O = P * V (V stored [B][C][N]); wave -> 48 channels ----
    {
        v8f acc[2][3];
        #pragma unroll
        for (int mt = 0; mt < 2; ++mt)
            #pragma unroll
            for (int nt = 0; nt < 3; ++nt) acc[mt][nt] = (v8f)(0.0f);

        for (int kc = 0; kc < Nt; kc += 32) {
            Frag16 A[2];
            #pragma unroll
            for (int mt = 0; mt < 2; ++mt) {
                const unsigned short* p = Ps + (mt * 16 + l15) * Nt + kc;
                A[mt].h[0] = *(const u32x4*)(p + half * 8);
                A[mt].h[1] = *(const u32x4*)(p + 16 + half * 8);
            }
            #pragma unroll
            for (int nt = 0; nt < 3; ++nt) {
                const int ch = wave * 48 + nt * 16 + l15;
                const unsigned short* p =
                    vT + ((size_t)(b * Ct + ch)) * Nt + kc;
                Frag16 Bf;
                Bf.h[0] = *(const u32x4*)(p + half * 8);
                Bf.h[1] = *(const u32x4*)(p + 16 + half * 8);
                acc[0][nt] = wmma_bf16(A[0].v, Bf.v, acc[0][nt]);
                acc[1][nt] = wmma_bf16(A[1].v, Bf.v, acc[1][nt]);
            }
        }
        // out[b][ch][n] (NCHW flattened): n = q0 + m
        #pragma unroll
        for (int mt = 0; mt < 2; ++mt)
            #pragma unroll
            for (int nt = 0; nt < 3; ++nt) {
                const int ch = wave * 48 + nt * 16 + l15;
                float* op = out + ((size_t)(b * Ct + ch)) * Nt + q0 + mt * 16 + 8 * half;
                #pragma unroll
                for (int r = 0; r < 8; ++r) op[r] = acc[mt][nt][r];
            }
    }
}

// ================= host launcher =================
extern "C" void kernel_launch(void* const* d_in, const int* in_sizes, int n_in,
                              void* d_out, int out_size, void* d_ws, size_t ws_size,
                              hipStream_t stream) {
    const float* x     = (const float*)d_in[0];   // [B,C,H,W]
    const float* w_qkv = (const float*)d_in[1];   // [3C,C]
    const float* b_qkv = (const float*)d_in[2];   // [3C]
    const float* w_dw  = (const float*)d_in[3];   // [C,1,3,3]
    const float* b_dw  = (const float*)d_in[4];   // [C]
    float* out = (float*)d_out;                   // [B,C,H,W]

    unsigned char* ws = (unsigned char*)d_ws;
    unsigned short* xb = (unsigned short*)(ws + OFF_XB);
    unsigned short* wb = (unsigned short*)(ws + OFF_WB);
    unsigned short* qb = (unsigned short*)(ws + OFF_QB);
    float*          kf = (float*)(ws + OFF_KF);
    unsigned short* kb = (unsigned short*)(ws + OFF_KB);
    unsigned short* vT = (unsigned short*)(ws + OFF_VT);

    k_cvt_w<<<(OCt * Ct + 255) / 256, 256, 0, stream>>>(w_qkv, wb);
    k_xpose<<<(Bt * Ct * Nt + 255) / 256, 256, 0, stream>>>(x, xb);
    k_qkv_gemm<<<dim3(OCt / 64, Mt / 128), 256, 0, stream>>>(xb, wb, b_qkv, qb, kf, vT);
    k_dwconv_addk<<<(Bt * Ct * Nt + 255) / 256, 256, 0, stream>>>(x, w_dw, b_dw, kf, kb);
    k_attention<<<Bt * (Nt / QROWS), 256, SMEM_TOTAL, stream>>>(qb, kb, vT, out);
}
